// HybridMemory_62062277427227
// MI455X (gfx1250) — compile-verified, matrix operand order checked by reading, not run on hardware
//
#include <hip/hip_runtime.h>

#define B_   256
#define D_   256
#define N_   100000
#define C_   5000
#define CP_  5008          // C padded to multiple of 16 for WMMA tiles
#define TEMP_INV 20.0f     // 1 / 0.05
#define EPS_ 1e-6f

typedef __attribute__((ext_vector_type(16))) _Float16 v16h;
typedef __attribute__((ext_vector_type(8)))  _Float16 v8h;
typedef __attribute__((ext_vector_type(8)))  float    v8f;

// ---------------- zero-init int scratch (counts + cursor) -----------------
__global__ void k_zero_i32(int* __restrict__ p, int n) {
    int i = blockIdx.x * blockDim.x + threadIdx.x;
    if (i < n) p[i] = 0;
}

// ---------------- label histogram (100K atomics, trivial) -----------------
__global__ void k_hist(const int* __restrict__ labels, int* __restrict__ counts) {
    int i = blockIdx.x * blockDim.x + threadIdx.x;
    if (i < N_) atomicAdd(&counts[labels[i]], 1);
}

// ---------------- single-block exclusive scan over C=5000 -----------------
__global__ void k_scan(const int* __restrict__ counts, int* __restrict__ offsets) {
    __shared__ int part[256];
    const int t = threadIdx.x;
    const int CHUNK = (C_ + 255) / 256;       // 20
    const int base = t * CHUNK;
    int s = 0;
    for (int i = 0; i < CHUNK; ++i) {
        int idx = base + i;
        if (idx < C_) s += counts[idx];
    }
    part[t] = s;
    __syncthreads();
    if (t == 0) {                              // serial scan of 256 partials
        int run = 0;
        for (int i = 0; i < 256; ++i) { int v = part[i]; part[i] = run; run += v; }
    }
    __syncthreads();
    int run = part[t];
    for (int i = 0; i < CHUNK; ++i) {
        int idx = base + i;
        if (idx < C_) { offsets[idx] = run; run += counts[idx]; }
    }
}

// ---------------- scatter row-ids into per-class member lists -------------
__global__ void k_scatter(const int* __restrict__ labels,
                          const int* __restrict__ offsets,
                          int* __restrict__ cursor,
                          int* __restrict__ member) {
    int i = blockIdx.x * blockDim.x + threadIdx.x;
    if (i < N_) {
        int lab = labels[i];
        int pos = atomicAdd(&cursor[lab], 1);
        member[offsets[lab] + pos] = i;
    }
}

// ---------------- atomic-free gather: Fagg[c,:] = sum of member rows ------
// one block per class; 256 threads = D columns; each member row is a single
// contiguous 1KB coalesced read. Non-temporal: the bank has zero reuse, so
// don't let this 102 MB stream churn L2 (the WMMA operands live there).
__global__ void k_gather(const float* __restrict__ feat,
                         const int*   __restrict__ member,
                         const int*   __restrict__ offsets,
                         const int*   __restrict__ counts,
                         float* __restrict__ Fagg) {
    int c = blockIdx.x;
    int d = threadIdx.x;
    int beg = offsets[c];
    int cnt = counts[c];                       // uniform per block
    float s = 0.0f;
    for (int i = 0; i < cnt; ++i) {
        int n = member[beg + i];
        s += __builtin_nontemporal_load(&feat[(size_t)n * D_ + d]);
    }
    Fagg[(size_t)c * D_ + d] = s;
}

// ---------------- convert inputs -> f16, folding 1/TEMP -------------------
__global__ void k_cvt_inputs(const float* __restrict__ x, _Float16* __restrict__ xh) {
    int i = blockIdx.x * blockDim.x + threadIdx.x;
    xh[i] = (_Float16)(x[i] * TEMP_INV);
}

// ---------------- convert F_agg -> f16, folding 1/counts, pad rows = 0 ----
__global__ void k_cvt_bank(const float* __restrict__ Fagg,
                           const int*   __restrict__ counts,
                           _Float16* __restrict__ gh) {
    int c = blockIdx.x;     // 0..CP_-1
    int d = threadIdx.x;    // 0..D_-1
    float v = 0.0f;
    if (c < C_) {
        int cnt = counts[c];
        float inv = (cnt > 0) ? (1.0f / (float)cnt) : 0.0f;
        v = Fagg[(size_t)c * D_ + d] * inv;
    }
    gh[(size_t)c * D_ + d] = (_Float16)v;
}

// ---------------- WMMA GEMM: sim[B,CP] = Xh[B,D] * Gh[CP,D]^T -------------
// one 16x16 output tile per wave32; K=256 -> 8 x v_wmma_f32_16x16x32_f16.
// A-frag (16x32 f16): lanes 0-15 row M=lane, halves K=k0+0..7 / k0+16..23;
// lanes 16-31 same rows, K=k0+8..15 / k0+24..31. B mirrors A (both row-major
// along K), so the per-lane load pattern is identical for both operands.
__global__ void k_wmma_gemm(const _Float16* __restrict__ Xh,
                            const _Float16* __restrict__ Gh,
                            float* __restrict__ sim) {
    const int wave = threadIdx.x >> 5;            // 8 waves per block
    const int lane = threadIdx.x & 31;
    const int tile = blockIdx.x * 8 + wave;       // 5008 tiles total
    const int m0 = (tile & 15) * 16;              // 16 tiles along B
    const int n0 = (tile >> 4) * 16;              // 313 tiles along CP
    const int hl = lane & 15;
    const int hi = lane >> 4;                     // 0 or 1

    const _Float16* aRow = Xh + (size_t)(m0 + hl) * D_ + hi * 8;
    const _Float16* bRow = Gh + (size_t)(n0 + hl) * D_ + hi * 8;

    v8f acc = {};
#pragma unroll
    for (int ks = 0; ks < D_; ks += 32) {
        v8h a0 = *(const v8h*)(aRow + ks);        // 16B global_load_b128
        v8h a1 = *(const v8h*)(aRow + ks + 16);
        v8h b0 = *(const v8h*)(bRow + ks);
        v8h b1 = *(const v8h*)(bRow + ks + 16);
        v16h A, Bf;
#pragma unroll
        for (int i = 0; i < 8; ++i) {
            A[i] = a0[i];  A[8 + i] = a1[i];
            Bf[i] = b0[i]; Bf[8 + i] = b1[i];
        }
        acc = __builtin_amdgcn_wmma_f32_16x16x32_f16(
            /*neg_a=*/false, A, /*neg_b=*/false, Bf,
            /*c_mod=*/(short)0, acc, /*reuse_a=*/false, /*reuse_b=*/false);
    }

    // C/D layout: VGPR r -> (M = m0+r+8*hi, N = n0+hl)
    float* out = sim + n0 + hl;
#pragma unroll
    for (int r = 0; r < 8; ++r) {
        out[(size_t)(m0 + r + hi * 8) * CP_] = acc[r];
    }
}

// ---------------- masked softmax + NLL per query --------------------------
__global__ void k_loss(const float* __restrict__ sim,
                       const int*   __restrict__ counts,
                       const int*   __restrict__ indexes,
                       const int*   __restrict__ labels,
                       float* __restrict__ partial) {
    __shared__ float red[256];
    int b = blockIdx.x;
    int t = threadIdx.x;
    const float* row = sim + (size_t)b * CP_;
    float s = 0.0f;
    for (int c = t; c < C_; c += 256) {
        if (counts[c] > 0) s += expf(row[c]);
    }
    red[t] = s;
    __syncthreads();
    for (int off = 128; off > 0; off >>= 1) {
        if (t < off) red[t] += red[t + off];
        __syncthreads();
    }
    if (t == 0) {
        int tgt = labels[indexes[b]];             // counts[tgt] >= 1 always
        float et = expf(row[tgt]);
        float prob = et / (red[0] + EPS_);
        partial[b] = -logf(prob + EPS_);
    }
}

// ---------------- mean over B ---------------------------------------------
__global__ void k_final(const float* __restrict__ partial, float* __restrict__ out) {
    __shared__ float red[256];
    int t = threadIdx.x;
    red[t] = partial[t];
    __syncthreads();
    for (int off = 128; off > 0; off >>= 1) {
        if (t < off) red[t] += red[t + off];
        __syncthreads();
    }
    if (t == 0) out[0] = red[0] / (float)B_;
}

extern "C" void kernel_launch(void* const* d_in, const int* in_sizes, int n_in,
                              void* d_out, int out_size, void* d_ws, size_t ws_size,
                              hipStream_t stream) {
    const float* inputs   = (const float*)d_in[0];   // [B,D] f32
    const int*   indexes  = (const int*)  d_in[1];   // [B]
    const float* features = (const float*)d_in[2];   // [N,D] f32
    const int*   labels   = (const int*)  d_in[3];   // [N]

    // workspace carve-up (256B aligned)
    char* ws = (char*)d_ws;
    size_t off = 0;
    auto carve = [&](size_t bytes) {
        char* p = ws + off;
        off = (off + bytes + 255) & ~(size_t)255;
        return p;
    };
    int*      cc      = (int*)     carve((size_t)2 * C_ * sizeof(int));   // counts|cursor
    int*      offsets = (int*)     carve((size_t)C_  * sizeof(int));
    int*      member  = (int*)     carve((size_t)N_  * sizeof(int));
    float*    Fagg    = (float*)   carve((size_t)C_  * D_ * sizeof(float));
    _Float16* Xh      = (_Float16*)carve((size_t)B_  * D_ * sizeof(_Float16));
    _Float16* Gh      = (_Float16*)carve((size_t)CP_ * D_ * sizeof(_Float16));
    float*    sim     = (float*)   carve((size_t)B_  * CP_ * sizeof(float));
    float*    partial = (float*)   carve((size_t)B_  * sizeof(float));
    int* counts = cc;
    int* cursor = cc + C_;

    const int NBLK = (N_ + 255) / 256;

    // 1) counting sort of rows by label (replaces 25.6M f32 atomics w/ 200K int)
    k_zero_i32<<<(2 * C_ + 255) / 256, 256, 0, stream>>>(cc, 2 * C_);
    k_hist   <<<NBLK, 256, 0, stream>>>(labels, counts);
    k_scan   <<<1, 256, 0, stream>>>(counts, offsets);
    k_scatter<<<NBLK, 256, 0, stream>>>(labels, offsets, cursor, member);

    // 2) atomic-free gather of the 102 MB bank (HBM-roofline kernel)
    k_gather<<<C_, D_, 0, stream>>>(features, member, offsets, counts, Fagg);

    // 3) precision fold-ins: X/TEMP -> f16, F_agg/counts -> f16 (padded)
    k_cvt_inputs<<<(B_ * D_) / 256, 256, 0, stream>>>(inputs, Xh);
    k_cvt_bank<<<CP_, D_, 0, stream>>>(Fagg, counts, Gh);

    // 4) 256 x 5008 x 256 GEMM on the WMMA path (8 waves/block, 1 tile/wave)
    k_wmma_gemm<<<(16 * (CP_ / 16)) / 8, 256, 0, stream>>>(Xh, Gh, sim);

    // 5) masked softmax + NLL per query, then mean
    k_loss<<<B_, 256, 0, stream>>>(sim, counts, indexes, labels, partial);
    k_final<<<1, 256, 0, stream>>>(partial, (float*)d_out);
}
